// CirculantMultiHeadAttention_21844203668314
// MI455X (gfx1250) — matrix operand order repaired and checked by reading, not compiled, not run on hardware
//
#include <hip/hip_runtime.h>
#include <hip/hip_bf16.h>

typedef __attribute__((ext_vector_type(16))) __bf16 v16bf;
typedef __attribute__((ext_vector_type(8)))  float  v8f;
typedef int v4i_t __attribute__((vector_size(16)));   // matches builtin param type

#define D_MODEL 1024
#define NUM_HEADS 16
#define HEAD_DIM 64
#define BATCH 4
#define SEQ 2048
#define ROWS (BATCH * SEQ) /* 8192 */

union Frag16 { v16bf v; unsigned int u[8]; };
union AccF   { v8f v; float f[8]; };

#ifndef __has_builtin
#define __has_builtin(x) 0
#endif
#if __has_builtin(__builtin_amdgcn_global_load_async_to_lds_b128)
#define HAVE_ASYNC_LDS 1
#else
#define HAVE_ASYNC_LDS 0
#endif

__device__ __forceinline__ unsigned short f2bf(float x) {
    unsigned int u = __float_as_uint(x);
    u += 0x7FFFu + ((u >> 16) & 1u);          // round-to-nearest-even
    return (unsigned short)(u >> 16);
}

// 16B global -> LDS copy: async DMA (ASYNCcnt) when available, sync fallback.
__device__ __forceinline__ void cp16(unsigned short* lds, const unsigned short* g) {
#if HAVE_ASYNC_LDS
    __builtin_amdgcn_global_load_async_to_lds_b128(
        (__attribute__((address_space(1))) v4i_t*)(unsigned long long)(const void*)g,
        (__attribute__((address_space(3))) v4i_t*)(unsigned int)(unsigned long long)(const void*)lds,
        0, 0);
#else
    *(uint4*)lds = *(const uint4*)g;
#endif
}

__device__ __forceinline__ void wait_async0() {
#if HAVE_ASYNC_LDS
#if __has_builtin(__builtin_amdgcn_s_wait_asynccnt)
    __builtin_amdgcn_s_wait_asynccnt(0);
#else
    asm volatile("s_wait_asynccnt 0x0" ::: "memory");
#endif
#endif
}

// ---------------- prep kernels ----------------
__global__ void k_cast_bf16(const float* __restrict__ src,
                            unsigned short* __restrict__ dst, int n) {
    int i = blockIdx.x * blockDim.x + threadIdx.x;
    int stride = gridDim.x * blockDim.x;
    for (; i < n; i += stride) dst[i] = f2bf(src[i]);
}

// Transposed circulant weights: Wt[n][k] = W[k][n] = c[(n - k) mod 1024].
__global__ void k_build_circ_t(const float* __restrict__ c,
                               unsigned short* __restrict__ Wt) {
    int i = blockIdx.x * blockDim.x + threadIdx.x;   // exactly 1M threads
    int n = i >> 10, k = i & 1023;
    Wt[i] = f2bf(c[(n - k) & 1023]);
}

// ------- bf16 WMMA GEMM:  Y = A(8192x1024) * W(1024x1024) + bias ----------
// Wt is the TRANSPOSED weight (Wt[n][k] = W[k][n]).
#define BM 128
#define BN 128
#define BK 32
#define KITERS (D_MODEL / BK)   /* 32 */

__global__ __launch_bounds__(256)
void k_gemm(const unsigned short* __restrict__ A,
            const unsigned short* __restrict__ Wt,
            const float* __restrict__ bias, float scale,
            unsigned short* __restrict__ outB,
            float* __restrict__ outF, int writeF32) {
    __shared__ unsigned short Alds[2][BM * BK];   // [row][k], double-buffered
    __shared__ unsigned short Blds[2][BN * BK];   // [n][k],   double-buffered
    const int tid = threadIdx.x;
    const int lane = tid & 31;
    const int wid = tid >> 5;
    const int wm = wid >> 1;                   // 0..3  (32-row slab)
    const int wn = wid & 1;                    // 0..1  (64-col slab)
    const int rowBase = blockIdx.y * BM;
    const int colBase = blockIdx.x * BN;
    const int mlane = lane & 15;
    const int hf = lane >> 4;

    AccF acc[2][4];
    #pragma unroll
    for (int i = 0; i < 2; i++)
        #pragma unroll
        for (int j = 0; j < 4; j++)
            #pragma unroll
            for (int r = 0; r < 8; r++) acc[i][j].f[r] = 0.f;

    // per-stage: 128x32 tile = 512 vec8, 2 vectors per thread per matrix
#if HAVE_ASYNC_LDS
    #pragma unroll
    for (int i = 0; i < 2; i++) {
        int vid = tid + i * 256;
        int r = vid >> 2;
        int cv = (vid & 3) * 8;
        cp16(&Alds[0][r * BK + cv], A  + (size_t)(rowBase + r) * D_MODEL + cv);
        cp16(&Blds[0][r * BK + cv], Wt + (size_t)(colBase + r) * D_MODEL + cv);
    }
#endif

    for (int it = 0; it < KITERS; ++it) {
        const int buf = it & 1;
#if HAVE_ASYNC_LDS
        wait_async0();          // this tile's DMA complete (my wave's ops)
        __syncthreads();        // all waves' DMA landed; prev compute done
        if (it + 1 < KITERS) {  // DMA next tile into other buffer, overlaps WMMA
            int k0n = (it + 1) * BK;
            #pragma unroll
            for (int i = 0; i < 2; i++) {
                int vid = tid + i * 256;
                int r = vid >> 2;
                int cv = (vid & 3) * 8;
                cp16(&Alds[buf ^ 1][r * BK + cv], A  + (size_t)(rowBase + r) * D_MODEL + k0n + cv);
                cp16(&Blds[buf ^ 1][r * BK + cv], Wt + (size_t)(colBase + r) * D_MODEL + k0n + cv);
            }
        }
#else
        __syncthreads();
        {
            int k0 = it * BK;
            #pragma unroll
            for (int i = 0; i < 2; i++) {
                int vid = tid + i * 256;
                int r = vid >> 2;
                int cv = (vid & 3) * 8;
                cp16(&Alds[buf][r * BK + cv], A  + (size_t)(rowBase + r) * D_MODEL + k0 + cv);
                cp16(&Blds[buf][r * BK + cv], Wt + (size_t)(colBase + r) * D_MODEL + k0 + cv);
            }
        }
        __syncthreads();
#endif

        Frag16 af[2], bfm[4];
        #pragma unroll
        for (int mi = 0; mi < 2; mi++) {
            int rA = wm * 32 + mi * 16 + mlane;
            #pragma unroll
            for (int j = 0; j < 8; j++) {
                int kk = 2 * j + 16 * hf;
                af[mi].u[j] = *(const unsigned int*)(&Alds[buf][rA * BK + kk]);
            }
        }
        #pragma unroll
        for (int ni = 0; ni < 4; ni++) {
            int cB = wn * 64 + ni * 16 + mlane;
            #pragma unroll
            for (int j = 0; j < 8; j++) {
                int kk = 2 * j + 16 * hf;
                bfm[ni].u[j] = *(const unsigned int*)(&Blds[buf][cB * BK + kk]);
            }
        }
        #pragma unroll
        for (int mi = 0; mi < 2; mi++)
            #pragma unroll
            for (int ni = 0; ni < 4; ni++)
                acc[mi][ni].v = __builtin_amdgcn_wmma_f32_16x16x32_bf16(
                    false, af[mi].v, false, bfm[ni].v, (short)0, acc[mi][ni].v, false, false);
    }

    if (writeF32) {
        #pragma unroll
        for (int mi = 0; mi < 2; mi++)
            #pragma unroll
            for (int ni = 0; ni < 4; ni++) {
                int gc = colBase + wn * 64 + ni * 16 + mlane;
                float bv = bias[gc];
                #pragma unroll
                for (int r = 0; r < 8; r++) {
                    int gr = rowBase + wm * 32 + mi * 16 + r + 8 * hf;
                    outF[(size_t)gr * D_MODEL + gc] = (acc[mi][ni].f[r] + bv) * scale;
                }
            }
    } else {
        #pragma unroll
        for (int mi = 0; mi < 2; mi++)
            #pragma unroll
            for (int ni = 0; ni < 4; ni++) {
                int gc = colBase + wn * 64 + ni * 16 + mlane;
                float bv = bias[gc];
                #pragma unroll
                for (int r = 0; r < 8; r++) {
                    int gr = rowBase + wm * 32 + mi * 16 + r + 8 * hf;
                    outB[(size_t)gr * D_MODEL + gc] = f2bf((acc[mi][ni].f[r] + bv) * scale);
                }
            }
    }
}

// ---------------- flash attention (bf16 WMMA, online softmax) ---------------
__global__ __launch_bounds__(256)
void k_attn(const unsigned short* __restrict__ Q,   // prescaled by 1/sqrt(dh)
            const unsigned short* __restrict__ K,
            const unsigned short* __restrict__ V,
            unsigned short* __restrict__ O) {
    __shared__ unsigned short Klds[64 * 64];        // [kv][d]
    __shared__ unsigned short Vt[64 * 64];          // [d][kv]
    __shared__ unsigned short Plds[8 * 16 * 64];    // per-wave P buffer
    const int tid = threadIdx.x;
    const int lane = tid & 31;
    const int wid = tid >> 5;
    const int mlane = lane & 15;
    const int hf = lane >> 4;
    const int b = blockIdx.y >> 4;
    const int h = blockIdx.y & 15;
    const int q0 = blockIdx.x * 128 + wid * 16;
    const size_t headOff = (size_t)h * HEAD_DIM;

    // Q A-fragments (2 k-steps over head dim)
    Frag16 qf[2];
    #pragma unroll
    for (int t = 0; t < 2; t++)
        #pragma unroll
        for (int j = 0; j < 8; j++) {
            int d0 = t * 32 + 2 * j + 16 * hf;
            qf[t].u[j] = *(const unsigned int*)(Q + (size_t)(b * SEQ + q0 + mlane) * D_MODEL + headOff + d0);
        }

    AccF oacc[4];
    #pragma unroll
    for (int j = 0; j < 4; j++)
        #pragma unroll
        for (int r = 0; r < 8; r++) oacc[j].f[r] = 0.f;
    float rowm[8], rowl[8];
    #pragma unroll
    for (int r = 0; r < 8; r++) { rowm[r] = -1e30f; rowl[r] = 0.f; }

    unsigned short* Pme = &Plds[wid * 16 * 64];

    for (int kv0 = 0; kv0 < SEQ; kv0 += 64) {
        __syncthreads();   // previous chunk's K/Vt reads complete
        // stage K (async DMA, row-major) and V (transposed, via regs)
        #pragma unroll
        for (int i = 0; i < 2; i++) {
            int vid = tid + i * 256;
            int r = vid >> 3;
            int c = (vid & 7) * 8;
            cp16(&Klds[r * 64 + c], K + (size_t)(b * SEQ + kv0 + r) * D_MODEL + headOff + c);
            uint4 vd = *(const uint4*)(V + (size_t)(b * SEQ + kv0 + r) * D_MODEL + headOff + c);
            const unsigned short* vs = (const unsigned short*)&vd;
            #pragma unroll
            for (int e = 0; e < 8; e++) Vt[(c + e) * 64 + r] = vs[e];
        }
        wait_async0();
        __syncthreads();

        // S = Q * K^T : 4 kv-subtiles of 16
        AccF sacc[4];
        #pragma unroll
        for (int j = 0; j < 4; j++) {
            #pragma unroll
            for (int r = 0; r < 8; r++) sacc[j].f[r] = 0.f;
            #pragma unroll
            for (int t = 0; t < 2; t++) {
                Frag16 kf;
                #pragma unroll
                for (int jj = 0; jj < 8; jj++) {
                    int dd = t * 32 + 2 * jj + 16 * hf;
                    kf.u[jj] = *(const unsigned int*)(&Klds[(j * 16 + mlane) * 64 + dd]);
                }
                sacc[j].v = __builtin_amdgcn_wmma_f32_16x16x32_bf16(
                    false, qf[t].v, false, kf.v, (short)0, sacc[j].v, false, false);
            }
        }

        // online softmax: rows r+8*hf, cols j*16+mlane
        #pragma unroll
        for (int r = 0; r < 8; r++) {
            float mc = fmaxf(fmaxf(sacc[0].f[r], sacc[1].f[r]),
                             fmaxf(sacc[2].f[r], sacc[3].f[r]));
            #pragma unroll
            for (int m = 1; m < 16; m <<= 1) mc = fmaxf(mc, __shfl_xor(mc, m, 32));
            float mnew = fmaxf(rowm[r], mc);
            float alpha = __expf(rowm[r] - mnew);
            rowm[r] = mnew;
            float psum = 0.f;
            #pragma unroll
            for (int j = 0; j < 4; j++) {
                float p = __expf(sacc[j].f[r] - mnew);
                psum += p;
                Pme[(r + 8 * hf) * 64 + j * 16 + mlane] = f2bf(p);
            }
            #pragma unroll
            for (int m = 1; m < 16; m <<= 1) psum += __shfl_xor(psum, m, 32);
            rowl[r] = rowl[r] * alpha + psum;
            #pragma unroll
            for (int j = 0; j < 4; j++) oacc[j].f[r] *= alpha;
        }
        asm volatile("s_wait_dscnt 0" ::: "memory");  // P: intra-wave cross-lane LDS RAW

        // O += P * V
        Frag16 pf[2];
        #pragma unroll
        for (int t = 0; t < 2; t++)
            #pragma unroll
            for (int j = 0; j < 8; j++) {
                int c0 = t * 32 + 2 * j + 16 * hf;
                pf[t].u[j] = *(const unsigned int*)(&Pme[mlane * 64 + c0]);
            }
        #pragma unroll
        for (int nd = 0; nd < 4; nd++)
            #pragma unroll
            for (int t = 0; t < 2; t++) {
                Frag16 vf;
                #pragma unroll
                for (int jj = 0; jj < 8; jj++) {
                    int kk = t * 32 + 2 * jj + 16 * hf;
                    vf.u[jj] = *(const unsigned int*)(&Vt[(nd * 16 + mlane) * 64 + kk]);
                }
                oacc[nd].v = __builtin_amdgcn_wmma_f32_16x16x32_bf16(
                    false, pf[t].v, false, vf.v, (short)0, oacc[nd].v, false, false);
            }
    }

    #pragma unroll
    for (int nd = 0; nd < 4; nd++)
        #pragma unroll
        for (int r = 0; r < 8; r++) {
            float val = oacc[nd].f[r] / rowl[r];
            int gr = q0 + r + 8 * hf;
            O[(size_t)(b * SEQ + gr) * D_MODEL + headOff + nd * 16 + mlane] = f2bf(val);
        }
}

// ---------------- launcher ----------------
extern "C" void kernel_launch(void* const* d_in, const int* in_sizes, int n_in,
                              void* d_out, int out_size, void* d_ws, size_t ws_size,
                              hipStream_t stream) {
    const float* x    = (const float*)d_in[0];
    const float* wq_c = (const float*)d_in[1];
    const float* wq_b = (const float*)d_in[2];
    const float* wk_c = (const float*)d_in[3];
    const float* wk_b = (const float*)d_in[4];
    const float* wv_c = (const float*)d_in[5];
    const float* wv_b = (const float*)d_in[6];
    const float* wo_c = (const float*)d_in[7];
    const float* wo_b = (const float*)d_in[8];
    float* out = (float*)d_out;

    char* ws = (char*)d_ws;
    size_t off = 0;
    auto alloc = [&](size_t bytes) -> char* {
        char* p = ws + off;
        off += (bytes + 255) & ~(size_t)255;
        return p;
    };
    unsigned short* xb = (unsigned short*)alloc((size_t)ROWS * D_MODEL * 2);
    unsigned short* Wq = (unsigned short*)alloc((size_t)D_MODEL * D_MODEL * 2);
    unsigned short* Wk = (unsigned short*)alloc((size_t)D_MODEL * D_MODEL * 2);
    unsigned short* Wv = (unsigned short*)alloc((size_t)D_MODEL * D_MODEL * 2);
    unsigned short* Wo = (unsigned short*)alloc((size_t)D_MODEL * D_MODEL * 2);
    unsigned short* Qb = (unsigned short*)alloc((size_t)ROWS * D_MODEL * 2);
    unsigned short* Kb = (unsigned short*)alloc((size_t)ROWS * D_MODEL * 2);
    unsigned short* Vb = (unsigned short*)alloc((size_t)ROWS * D_MODEL * 2);
    unsigned short* Ab = (unsigned short*)alloc((size_t)ROWS * D_MODEL * 2);

    k_cast_bf16<<<4096, 256, 0, stream>>>(x, xb, ROWS * D_MODEL);
    k_build_circ_t<<<4096, 256, 0, stream>>>(wq_c, Wq);
    k_build_circ_t<<<4096, 256, 0, stream>>>(wk_c, Wk);
    k_build_circ_t<<<4096, 256, 0, stream>>>(wv_c, Wv);
    k_build_circ_t<<<4096, 256, 0, stream>>>(wo_c, Wo);

    dim3 ggrid(D_MODEL / BN, ROWS / BM);
    const float qscale = 0.125f;  // 1/sqrt(HEAD_DIM), folded into Q
    k_gemm<<<ggrid, 256, 0, stream>>>(xb, Wq, wq_b, qscale, Qb, nullptr, 0);
    k_gemm<<<ggrid, 256, 0, stream>>>(xb, Wk, wk_b, 1.0f,   Kb, nullptr, 0);
    k_gemm<<<ggrid, 256, 0, stream>>>(xb, Wv, wv_b, 1.0f,   Vb, nullptr, 0);

    k_attn<<<dim3(SEQ / 128, BATCH * NUM_HEADS), 256, 0, stream>>>(Qb, Kb, Vb, Ab);

    k_gemm<<<ggrid, 256, 0, stream>>>(Ab, Wo, wo_b, 1.0f, nullptr, out, 1);
}
